// AblationDecoder_New_67654324846834
// MI455X (gfx1250) — matrix operand order repaired and checked by reading, not compiled, not run on hardware
//
#include <hip/hip_runtime.h>

typedef _Float16 h2   __attribute__((ext_vector_type(2)));
typedef _Float16 v16h __attribute__((ext_vector_type(16)));
typedef float    v8f  __attribute__((ext_vector_type(8)));
typedef float    f4   __attribute__((ext_vector_type(4)));

#define NPTS   50000
#define NTILES (NPTS / 16)   // 3125 exactly

// ---------------------------------------------------------------------------
// helpers
// ---------------------------------------------------------------------------
static __device__ __forceinline__ h2 pkh(float a, float b) {
  h2 r; r.x = (_Float16)a; r.y = (_Float16)b; return r;
}
static __device__ __forceinline__ int pkrelu(float a, float b) {
  union { h2 h; int i; } u;
  u.h.x = (_Float16)fmaxf(a, 0.f);
  u.h.y = (_Float16)fmaxf(b, 0.f);
  return u.i;
}
static __device__ __forceinline__ v8f WMMA(v16h a, v16h b, v8f c) {
  // D(16x16 f32) = A(16x32 f16) * B(32x16 f16) + C
  return __builtin_amdgcn_wmma_f32_16x16x32_f16(false, a, false, b, (short)0, c,
                                                false, false);
}
// Pre-swizzled weight A-fragment. wfl = wf + lane*16, so with compile-time
// (g,m,kf) the whole index is an instruction-immediate offset.
static __device__ __forceinline__ v16h afrag(const _Float16* __restrict__ wfl,
                                             int g, int m, int kf) {
  return *(const v16h*)(wfl + (size_t)(((g * 4 + m) * 2 + kf) * 512));
}
// Build B-fragment (32x16, K=features, N=points) from two C/D tiles of netT.
// C/D layout: lane<16 -> N=lane, VGPR v -> M=v ; lane>=16 -> N=lane-16, M=8+v.
// B layout:   lane<16 -> K=2j,2j+1 ; lane>=16 -> K=16+2j,16+2j+1 (N=lane%16).
// => half-wave swap (SWAPX16 ds_swizzle, offset 0x401f) moves M=8..15 between halves.
static __device__ __forceinline__ v16h make_bfrag(v8f T0, v8f T1, bool lo) {
  int Pa[4], Pb[4];
#pragma unroll
  for (int j = 0; j < 4; ++j) {
    Pa[j] = pkrelu(T0[2 * j], T0[2 * j + 1]);
    Pb[j] = pkrelu(T1[2 * j], T1[2 * j + 1]);
  }
  union { v16h v; int u[8]; } r;
#pragma unroll
  for (int j = 0; j < 4; ++j) {
    int Sa = __builtin_amdgcn_ds_swizzle(Pa[j], 0x401f);  // lane ^= 16
    int Sb = __builtin_amdgcn_ds_swizzle(Pb[j], 0x401f);
    r.u[j]     = lo ? Pa[j] : Sb;
    r.u[j + 4] = lo ? Sa    : Pb[j];
  }
  return r.v;
}
// B-fragment straight from global f32 rows of c (no relu).
static __device__ __forceinline__ v16h cfrag(const float* __restrict__ crow,
                                             int kf, int hi) {
  const f4* s = (const f4*)(crow + kf * 32 + hi * 16);
  f4 a = s[0], b = s[1], cc = s[2], d = s[3];
  union { v16h v; h2 h[8]; } u;
  u.h[0] = pkh(a[0], a[1]);   u.h[1] = pkh(a[2], a[3]);
  u.h[2] = pkh(b[0], b[1]);   u.h[3] = pkh(b[2], b[3]);
  u.h[4] = pkh(cc[0], cc[1]); u.h[5] = pkh(cc[2], cc[3]);
  u.h[6] = pkh(d[0], d[1]);   u.h[7] = pkh(d[2], d[3]);
  return u.v;
}

// ---------------------------------------------------------------------------
// prep 1: per-batch conditioning bias  tf @ fc_c_W[i][64:] + fc_c_b[i]
// ---------------------------------------------------------------------------
__global__ void prep_bias(const float* __restrict__ tf, const float* __restrict__ fc_c_W,
                          const float* __restrict__ fc_c_b, float* __restrict__ cbias) {
  int b = blockIdx.x, i = blockIdx.y, o = threadIdx.x;  // grid(8,5) block(64)
  float s = fc_c_b[i * 64 + o];
  const float* w = fc_c_W + ((size_t)i * 576 + 64) * 64 + o;
  const float* t = tf + b * 512;
  for (int k = 0; k < 512; ++k) s += t[k] * w[(size_t)k * 64];
  cbias[(b * 5 + i) * 64 + o] = s;
}

// ---------------------------------------------------------------------------
// prep 2: weight A-fragments (W^T as 16x32 A frags, contiguous k = 16*(lane/16) + j), f16.
// slots g: 3i+0 = fc_c (c-part), 3i+1 = blk0, 3i+2 = blk1, 15 = fc_out(pad16),
//          16 = fc_p (K padded to 32; K=3 row carries fc_p_b for bias fold)
// ---------------------------------------------------------------------------
__global__ void prep_wfrags(const float* __restrict__ fc_c_W, const float* __restrict__ blk0_W,
                            const float* __restrict__ blk1_W, const float* __restrict__ fc_out_W,
                            const float* __restrict__ fc_p_W, const float* __restrict__ fc_p_b,
                            _Float16* __restrict__ wf) {
  int g = blockIdx.x;                       // 0..16
  int t = threadIdx.x;                      // 256
  int m = t >> 6, kf = (t >> 5) & 1, lane = t & 31;
  int o  = m * 16 + (lane & 15);            // output feature (A-matrix M)
  int hi = lane >> 4;
  _Float16* dst = wf + (size_t)((((g * 4 + m) * 2 + kf) * 32 + lane) * 16);
#pragma unroll
  for (int j = 0; j < 16; ++j) {
    // A layout (contiguous, matches B): lane<16: K = j; lane>=16: K = 16 + j
    int K = kf * 32 + hi * 16 + j;
    float v = 0.f;
    if (g < 15) {
      int i = g / 3, kind = g - 3 * i;
      if (kind == 0)      v = fc_c_W[((size_t)i * 576 + K) * 64 + o];
      else if (kind == 1) v = blk0_W[((size_t)i * 64 + K) * 64 + o];
      else                v = blk1_W[((size_t)i * 64 + K) * 64 + o];
    } else if (g == 15) {
      if (m == 0 && o < 12) v = fc_out_W[K * 12 + o];
    } else {  // g == 16 : fc_p, only kf==0 non-zero
      if (kf == 0) {
        if (K < 3)       v = fc_p_W[K * 64 + o];
        else if (K == 3) v = fc_p_b[o];     // bias via ones-column trick
      }
    }
    dst[j] = (_Float16)v;
  }
}

// ---------------------------------------------------------------------------
// main: one 16-point tile per wave, netT (64 feat x 16 pts) = 4 WMMA acc tiles
// ---------------------------------------------------------------------------
__global__ __launch_bounds__(256) void decoder_main(
    const float* __restrict__ p, const float* __restrict__ c,
    const float* __restrict__ cbias, const _Float16* __restrict__ wf,
    const float* __restrict__ b0g, const float* __restrict__ b1g,
    const float* __restrict__ bog, float* __restrict__ out) {
  __shared__ float lds[976];  // [0,320) cond bias, [320,640) b0, [640,960) b1, [960,976) bout
  const int tid = threadIdx.x;
  const int b = blockIdx.y;
  for (int idx = tid; idx < 976; idx += 256) {
    float v;
    if (idx < 320)      v = cbias[b * 320 + idx];
    else if (idx < 640) v = b0g[idx - 320];
    else if (idx < 960) v = b1g[idx - 640];
    else                v = (idx < 972) ? bog[idx - 960] : 0.f;
    lds[idx] = v;
  }
  __syncthreads();

  const int wave = tid >> 5;
  const int lane = tid & 31;
  const int tile = blockIdx.x * 8 + wave;
  if (tile >= NTILES) return;              // wave-uniform; no barriers after this

  const int pt0  = tile * 16;
  const bool lo  = lane < 16;
  const int hi   = lane >> 4;
  const int mrow = lane & 15;
  const v8f zc = {0.f, 0.f, 0.f, 0.f, 0.f, 0.f, 0.f, 0.f};

  // per-lane bases: all fragment/bias offsets below are compile-time immediates
  const _Float16* __restrict__ wfl = wf + lane * 16;
  const float* __restrict__ lb = lds + hi * 8;

  // ---- net = p @ fc_p_W + fc_p_b  (K padded to 32, bias via ones at K=3) ----
  const float* prow = p + ((size_t)b * NPTS + pt0 + mrow) * 3;
  float px = prow[0], py = prow[1], pz = prow[2];
  h2 z2 = pkh(0.f, 0.f);
  union { v16h v; h2 h[8]; } bp;
#pragma unroll
  for (int j = 0; j < 8; ++j) bp.h[j] = z2;
  bp.h[0] = lo ? pkh(px, py)  : z2;        // K=0,1
  bp.h[1] = lo ? pkh(pz, 1.f) : z2;        // K=2, K=3(=1 -> bias row)

  v8f net[4];
#pragma unroll
  for (int m = 0; m < 4; ++m)
    net[m] = WMMA(afrag(wfl, 16, m, 0), bp.v, zc);

  const float* crow = c + ((size_t)b * NPTS + pt0 + mrow) * 64;

#pragma unroll
  for (int i = 0; i < 5; ++i) {
    // conditioning: net += Wc^T x c^T + (tf-bias + fc_c_b)
    v16h c0 = cfrag(crow, 0, hi);
    v16h c1 = cfrag(crow, 1, hi);
#pragma unroll
    for (int m = 0; m < 4; ++m) {
      net[m] = WMMA(afrag(wfl, 3 * i + 0, m, 0), c0, net[m]);
      net[m] = WMMA(afrag(wfl, 3 * i + 0, m, 1), c1, net[m]);
    }
#pragma unroll
    for (int t = 0; t < 4; ++t)
#pragma unroll
      for (int v = 0; v < 8; ++v)
        net[t][v] += lb[i * 64 + t * 16 + v];

    // blk0: h = relu(net) @ W0 + b0
    v16h f0 = make_bfrag(net[0], net[1], lo);
    v16h f1 = make_bfrag(net[2], net[3], lo);
    v8f hh[4];
#pragma unroll
    for (int m = 0; m < 4; ++m) {
      hh[m] = WMMA(afrag(wfl, 3 * i + 1, m, 0), f0, zc);
      hh[m] = WMMA(afrag(wfl, 3 * i + 1, m, 1), f1, hh[m]);
    }
#pragma unroll
    for (int t = 0; t < 4; ++t)
#pragma unroll
      for (int v = 0; v < 8; ++v)
        hh[t][v] += lb[320 + i * 64 + t * 16 + v];

    // blk1 + residual: net += relu(h) @ W1 + b1   (residual folded into C)
    f0 = make_bfrag(hh[0], hh[1], lo);
    f1 = make_bfrag(hh[2], hh[3], lo);
#pragma unroll
    for (int m = 0; m < 4; ++m) {
      net[m] = WMMA(afrag(wfl, 3 * i + 2, m, 0), f0, net[m]);
      net[m] = WMMA(afrag(wfl, 3 * i + 2, m, 1), f1, net[m]);
    }
#pragma unroll
    for (int t = 0; t < 4; ++t)
#pragma unroll
      for (int v = 0; v < 8; ++v)
        net[t][v] += lb[640 + i * 64 + t * 16 + v];
  }

  // out = relu(net) @ fc_out_W + fc_out_b   (Wout^T zero-padded to 16 rows)
  v16h g0 = make_bfrag(net[0], net[1], lo);
  v16h g1 = make_bfrag(net[2], net[3], lo);
  v8f o = WMMA(afrag(wfl, 15, 0, 0), g0, zc);
  o     = WMMA(afrag(wfl, 15, 0, 1), g1, o);

  // D layout: lane<16 -> out-ch v (0..7); lane>=16 -> out-ch 8+v (valid v<4)
  float* orow = out + ((size_t)b * NPTS + pt0 + mrow) * 12;
  if (lo) {
    f4 o0 = { o[0] + lds[960], o[1] + lds[961], o[2] + lds[962], o[3] + lds[963] };
    f4 o1 = { o[4] + lds[964], o[5] + lds[965], o[6] + lds[966], o[7] + lds[967] };
    *(f4*)(orow)     = o0;
    *(f4*)(orow + 4) = o1;
  } else {
    f4 o2 = { o[0] + lds[968], o[1] + lds[969], o[2] + lds[970], o[3] + lds[971] };
    *(f4*)(orow + 8) = o2;
  }
}

// ---------------------------------------------------------------------------
extern "C" void kernel_launch(void* const* d_in, const int* in_sizes, int n_in,
                              void* d_out, int out_size, void* d_ws, size_t ws_size,
                              hipStream_t stream) {
  (void)in_sizes; (void)n_in; (void)out_size; (void)ws_size;
  const float* p        = (const float*)d_in[0];
  const float* c        = (const float*)d_in[1];
  const float* tf       = (const float*)d_in[2];
  const float* fc_p_W   = (const float*)d_in[3];
  const float* fc_p_b   = (const float*)d_in[4];
  const float* fc_c_W   = (const float*)d_in[5];
  const float* fc_c_b   = (const float*)d_in[6];
  const float* blk0_W   = (const float*)d_in[7];
  const float* blk0_b   = (const float*)d_in[8];
  const float* blk1_W   = (const float*)d_in[9];
  const float* blk1_b   = (const float*)d_in[10];
  const float* fc_out_W = (const float*)d_in[11];
  const float* fc_out_b = (const float*)d_in[12];
  float* out = (float*)d_out;

  // workspace: [0,10240) f32 cond-bias[8][5][64]; [10240, +139264) f16 A-frags
  float*    cbias = (float*)d_ws;
  _Float16* wf    = (_Float16*)((char*)d_ws + 10240);

  prep_bias<<<dim3(8, 5), 64, 0, stream>>>(tf, fc_c_W, fc_c_b, cbias);
  prep_wfrags<<<17, 256, 0, stream>>>(fc_c_W, blk0_W, blk1_W, fc_out_W,
                                      fc_p_W, fc_p_b, wf);
  decoder_main<<<dim3((NTILES + 7) / 8, 8), 256, 0, stream>>>(
      p, c, cbias, wf, blk0_b, blk1_b, fc_out_b, out);
}
